// MLA_59081570125085
// MI455X (gfx1250) — matrix operand order repaired and checked by reading, not compiled, not run on hardware
//
#include <hip/hip_runtime.h>
#include <hip/hip_bf16.h>
#include <math.h>

// ---------------------------------------------------------------------------
// MLA for gfx1250 (MI455X): bf16 WMMA pipeline, K-major B operands so every
// WMMA fragment loads as ds_load_b128, async global->LDS tile staging with
// double buffering in the GEMM, flash attention with online softmax.
// ---------------------------------------------------------------------------

typedef __attribute__((ext_vector_type(16))) __bf16 v16bf;
typedef __attribute__((ext_vector_type(8)))  float  v8f;
typedef int v4i __attribute__((vector_size(16)));

union BF16x16 { v16bf v; unsigned short u[16]; unsigned int w[8]; };

#if __has_builtin(__builtin_amdgcn_global_load_async_to_lds_b128)
#define ASYNC_G2L 1
#else
#define ASYNC_G2L 0
#endif

__device__ __forceinline__ void g2l_16B(const unsigned short* g,
                                        unsigned short* l) {
#if ASYNC_G2L
  __builtin_amdgcn_global_load_async_to_lds_b128(
      (__attribute__((address_space(1))) v4i*)(g),
      (__attribute__((address_space(3))) v4i*)(l), 0, 0);
#else
  *(uint4*)l = *(const uint4*)g;
#endif
}

__device__ __forceinline__ void async_wait0() {
#if ASYNC_G2L
#if __has_builtin(__builtin_amdgcn_s_wait_asynccnt)
  __builtin_amdgcn_s_wait_asynccnt(0);
#else
  asm volatile("s_wait_asynccnt 0x0" ::: "memory");
#endif
#endif
}

__device__ __forceinline__ unsigned short f2bf(float f) {
  unsigned int u = __builtin_bit_cast(unsigned int, f);
  u += 0x7fffu + ((u >> 16) & 1u);  // round-to-nearest-even
  return (unsigned short)(u >> 16);
}

__device__ __forceinline__ v8f vzero8() {
  v8f z;
#pragma unroll
  for (int i = 0; i < 8; ++i) z[i] = 0.0f;
  return z;
}

__device__ __forceinline__ v8f wmma_bf16(v16bf a, v16bf b, v8f c) {
  return __builtin_amdgcn_wmma_f32_16x16x32_bf16(
      false, a, false, b, (short)0, c, false, false);
}

// A-fragment (16x32 bf16) from LDS (row-major, K contiguous).
// lane L: row = L%16; vgpr p: K = (p>=4?16:0) + (L/16)*8 + (p&3)*2 (+1)
// -> two contiguous 16B chunks per lane => 2x ds_load_b128.
template <int STRIDE>
__device__ __forceinline__ v16bf lds_a_frag(const unsigned short (*s)[STRIDE],
                                            int row0, int k0, int lane) {
  BF16x16 f;
  const unsigned short* r = s[row0 + (lane & 15)];
  const int kh = (lane >> 4) << 3;
#pragma unroll
  for (int p = 0; p < 8; ++p) {
    const int k = k0 + ((p & 4) << 2) + kh + ((p & 3) << 1);
    f.w[p] = *(const unsigned int*)(&r[k]);
  }
  return f.v;
}

// B-fragment (32x16 bf16) from a K-major LDS tile s[col][k]:
// lane L: col = col0 + L%16; vgpr p: K = k0 + (p>=4?16:0) + (L/16)*8 + (p&3)*2 (+1)
// -> two contiguous 16B chunks per lane => 2x ds_load_b128.
template <int STRIDE>
__device__ __forceinline__ v16bf lds_bt_frag(const unsigned short (*s)[STRIDE],
                                             int col0, int k0, int lane) {
  BF16x16 f;
  const unsigned short* r = s[col0 + (lane & 15)] + k0 + ((lane >> 4) << 3);
#pragma unroll
  for (int p = 0; p < 8; ++p)
    f.w[p] = *(const unsigned int*)(&r[2 * p + ((p & 4) << 1)]);
  return f.v;
}

// A-fragment from global row-major [rows, ld]
__device__ __forceinline__ v16bf glb_a_frag(const unsigned short* base, int ld,
                                            int lane) {
  BF16x16 f;
  const unsigned short* r = base + (size_t)(lane & 15) * ld;
  const int kh = (lane >> 4) << 3;
#pragma unroll
  for (int p = 0; p < 8; ++p) {
    const int k = ((p & 4) << 2) + kh + ((p & 3) << 1);
    f.w[p] = *(const unsigned int*)(&r[k]);
  }
  return f.v;
}

// ---------------------------------------------------------------------------
// f32 -> bf16 cast (n multiple of 256)
// ---------------------------------------------------------------------------
__global__ void cast_f32_bf16_kernel(const float* __restrict__ in,
                                     unsigned short* __restrict__ out,
                                     size_t n) {
  size_t i = (size_t)blockIdx.x * blockDim.x + threadIdx.x;
  if (i < n) out[i] = f2bf(in[i]);
}

// f32 [K,N] -> bf16 [N,K] (cast + transpose, one-time weight prep)
__global__ void castT_f32_bf16_kernel(const float* __restrict__ in,
                                      unsigned short* __restrict__ out, int K,
                                      int N) {
  size_t i = (size_t)blockIdx.x * blockDim.x + threadIdx.x;
  if (i < (size_t)K * N) {
    int k = (int)(i / N), n = (int)(i % N);
    out[(size_t)n * K + k] = f2bf(in[i]);
  }
}

// ---------------------------------------------------------------------------
// bf16 GEMM: C[M,N] = A[M,K] @ B[K,N] + bias[N], B given K-major (BT=[N,K]).
// 128x128 block, K-step 32, 256 threads = 8 waves (2x4), wave = 64x32 region.
// Double-buffered LDS tiles fed by async global->LDS b128 copies.
// vt==1: write bf16 output transposed as V^T [B,NH,HD,S] (B=2,S=2048 fixed).
// ---------------------------------------------------------------------------
__global__ __launch_bounds__(256) void gemm_bf16_kernel(
    const unsigned short* __restrict__ A, int lda,
    const unsigned short* __restrict__ BT, int ldbt,
    const float* __restrict__ bias,
    float* __restrict__ Cf, unsigned short* __restrict__ Cb, int ldc, int vt,
    int M, int N, int K) {
  __shared__ unsigned short sA[2][128][40];   // [M rows][K], +8 pad
  __shared__ unsigned short sB[2][128][40];   // [N cols][K], +8 pad

  const int tid = threadIdx.x;
  const int lane = tid & 31;
  const int wid = tid >> 5;
  const int wm = wid >> 2, wn = wid & 3;
  const int m0 = blockIdx.y * 128, n0 = blockIdx.x * 128;

  v8f acc[4][2];
#pragma unroll
  for (int i = 0; i < 4; ++i)
#pragma unroll
    for (int j = 0; j < 2; ++j) acc[i][j] = vzero8();

  const int pr = tid >> 2, pc = (tid & 3) * 8;  // 64 rows per pass, 8 bf16 ea
  const int nkt = K >> 5;

  auto load_tile = [&](int kt, int buf) {
    const int k0 = kt << 5;
    g2l_16B(&A[(size_t)(m0 + pr) * lda + k0 + pc], &sA[buf][pr][pc]);
    g2l_16B(&A[(size_t)(m0 + pr + 64) * lda + k0 + pc], &sA[buf][pr + 64][pc]);
    g2l_16B(&BT[(size_t)(n0 + pr) * ldbt + k0 + pc], &sB[buf][pr][pc]);
    g2l_16B(&BT[(size_t)(n0 + pr + 64) * ldbt + k0 + pc],
            &sB[buf][pr + 64][pc]);
  };

  int cur = 0;
  load_tile(0, 0);
  async_wait0();
  __syncthreads();

  for (int kt = 0; kt < nkt; ++kt) {
    if (kt + 1 < nkt) load_tile(kt + 1, cur ^ 1);

    v16bf bf0 = lds_bt_frag<40>(sB[cur], wn * 32, 0, lane);
    v16bf bf1 = lds_bt_frag<40>(sB[cur], wn * 32 + 16, 0, lane);
#pragma unroll
    for (int mt = 0; mt < 4; ++mt) {
      v16bf af = lds_a_frag<40>(sA[cur], wm * 64 + mt * 16, 0, lane);
      acc[mt][0] = wmma_bf16(af, bf0, acc[mt][0]);
      acc[mt][1] = wmma_bf16(af, bf1, acc[mt][1]);
    }
    async_wait0();
    __syncthreads();
    cur ^= 1;
  }

  const int kh = lane >> 4, nl = lane & 15;
#pragma unroll
  for (int mt = 0; mt < 4; ++mt) {
#pragma unroll
    for (int nt = 0; nt < 2; ++nt) {
      const int col = n0 + wn * 32 + nt * 16 + nl;
      const float bv = bias ? bias[col] : 0.0f;
#pragma unroll
      for (int r = 0; r < 8; ++r) {
        const int row = m0 + wm * 64 + mt * 16 + r + 8 * kh;
        const float v = acc[mt][nt][r] + bv;
        if (Cf) {
          Cf[(size_t)row * ldc + col] = v;
        } else if (vt) {
          // V^T [B,NH,HD,S]: row=(b*2048+s), col=(h*128+d)
          Cb[(size_t)(row >> 11) * 4194304 + (size_t)col * 2048 +
             (row & 2047)] = f2bf(v);
        } else {
          Cb[(size_t)row * ldc + col] = f2bf(v);
        }
      }
    }
  }
}

// ---------------------------------------------------------------------------
// RoPE + pack into [B,NH,S,160] bf16 (dims 0-127 rope'd latent, 128-135 rope
// dims, 136-159 zero pad so attention K-loops are multiples of 32).
// ---------------------------------------------------------------------------
__global__ __launch_bounds__(128) void rope_pack_kernel(
    const float* __restrict__ lat,   // [B,S,NH,128]
    const float* __restrict__ rp,    // [B,S,NH,8]
    const float* __restrict__ cosc,  // [2048,128]
    const float* __restrict__ sinc,  // [2048,128]
    const int* __restrict__ pos,     // [S]
    unsigned short* __restrict__ out) {
  const int S = 2048, NH = 16, HD = 128, RD = 8, Dp = 160;
  const int s = blockIdx.x, b = blockIdx.y, d = threadIdx.x;
  const int p = pos[s];
  const float c = cosc[(size_t)p * HD + d];
  const float sn = sinc[(size_t)p * HD + d];

  for (int h = 0; h < NH; ++h) {
    const size_t ib = ((size_t)(b * S + s) * NH + h) * HD;
    const float x = lat[ib + d];
    const float rot = (d < 64) ? -lat[ib + d + 64] : lat[ib + d - 64];
    const size_t ob = ((size_t)(b * NH + h) * S + s) * Dp;
    out[ob + d] = f2bf(x * c + rot * sn);
    if (d < RD)
      out[ob + HD + d] = f2bf(rp[((size_t)(b * S + s) * NH + h) * RD + d]);
    if (d < Dp - HD - RD) out[ob + HD + RD + d] = 0;
  }
}

// ---------------------------------------------------------------------------
// Flash attention (causal, matches reference's -1e9 fill).
// Block = 128 queries x 1 head, 256 threads (8 waves, 16 query rows each).
// Q,K: [B,NH,S,160] bf16;  V^T: [B,NH,128,S] bf16;  O: [B,S,NH,128] bf16.
// K tile LDS is key-major [kv][d] (d contiguous -> b128 B-frags);
// V tile LDS is d-major [d][kv] (kv contiguous -> b128 B-frags).
// ---------------------------------------------------------------------------
__global__ __launch_bounds__(256) void mla_attn_kernel(
    const unsigned short* __restrict__ Qp,
    const unsigned short* __restrict__ Kp,
    const unsigned short* __restrict__ VT,
    unsigned short* __restrict__ Ob, float inv_scale) {
  const int S = 2048, NH = 16, Dp = 160, HD = 128;
  const int AQ = 128, AKV = 64;
  const int q0 = blockIdx.x * AQ;
  const int h = blockIdx.y, b = blockIdx.z;
  const int tid = threadIdx.x, lane = tid & 31, wid = tid >> 5;
  const int kh = lane >> 4, nl = lane & 15;

  __shared__ unsigned short sK[64][168];    // [kv][d 0..159] +8 pad
  __shared__ unsigned short sV[128][72];    // [d][kv 0..63] +8 pad
  __shared__ unsigned short sP[8][16][72];  // per-wave P relayout

  const unsigned short* qbase =
      Qp + ((size_t)(b * NH + h) * S + q0 + wid * 16) * Dp;
  v16bf qf[5];
#pragma unroll
  for (int ks = 0; ks < 5; ++ks) qf[ks] = glb_a_frag(qbase + ks * 32, Dp, lane);

  v8f o[8];
#pragma unroll
  for (int i = 0; i < 8; ++i) o[i] = vzero8();
  float mrow[8], lrow[8];
#pragma unroll
  for (int r = 0; r < 8; ++r) { mrow[r] = -1e30f; lrow[r] = 0.0f; }

  const unsigned short* kbase = Kp + (size_t)(b * NH + h) * S * Dp;
  const unsigned short* vbase = VT + (size_t)(b * NH + h) * HD * S;

  const int nkb = (q0 + AQ) / AKV;
  for (int kb = 0; kb < nkb; ++kb) {
    const int kvs = kb * AKV;
    // K tile: 64 rows x 160 d = 1280 b128 copies, 5 per thread
#pragma unroll
    for (int it = 0; it < 5; ++it) {
      const int idx = tid + it * 256;
      const int r = idx / 20, cg = (idx % 20) * 8;
      g2l_16B(&kbase[(size_t)(kvs + r) * Dp + cg], &sK[r][cg]);
    }
    // V^T tile: 128 rows x 64 kv = 1024 b128 copies, 4 per thread
#pragma unroll
    for (int it = 0; it < 4; ++it) {
      const int idx = tid + it * 256;
      const int d = idx >> 3, cg = (idx & 7) * 8;
      g2l_16B(&vbase[(size_t)d * S + kvs + cg], &sV[d][cg]);
    }
    async_wait0();
    __syncthreads();

    // S = Q K^T over 160 dims (5 k-steps), 4 key sub-tiles
    v8f sc[4];
#pragma unroll
    for (int nt = 0; nt < 4; ++nt) {
      sc[nt] = vzero8();
#pragma unroll
      for (int ks = 0; ks < 5; ++ks)
        sc[nt] = wmma_bf16(qf[ks], lds_bt_frag<168>(sK, nt * 16, ks * 32, lane),
                           sc[nt]);
    }

    // scale + causal mask + row max
    float xr[8];
#pragma unroll
    for (int r = 0; r < 8; ++r) xr[r] = -3.0e38f;
#pragma unroll
    for (int nt = 0; nt < 4; ++nt) {
      const int kg = kvs + nt * 16 + nl;
#pragma unroll
      for (int r = 0; r < 8; ++r) {
        const int qg = q0 + wid * 16 + r + 8 * kh;
        float v = sc[nt][r] * inv_scale;
        if (kg > qg) v = -1e9f;
        sc[nt][r] = v;
        xr[r] = fmaxf(xr[r], v);
      }
    }
#pragma unroll
    for (int r = 0; r < 8; ++r) {
      xr[r] = fmaxf(xr[r], __shfl_xor(xr[r], 1, 32));
      xr[r] = fmaxf(xr[r], __shfl_xor(xr[r], 2, 32));
      xr[r] = fmaxf(xr[r], __shfl_xor(xr[r], 4, 32));
      xr[r] = fmaxf(xr[r], __shfl_xor(xr[r], 8, 32));
    }
    float mnew[8], alpha[8], rs[8];
#pragma unroll
    for (int r = 0; r < 8; ++r) {
      mnew[r] = fmaxf(mrow[r], xr[r]);
      alpha[r] = __expf(mrow[r] - mnew[r]);
      rs[r] = 0.0f;
    }
#pragma unroll
    for (int nt = 0; nt < 4; ++nt) {
#pragma unroll
      for (int r = 0; r < 8; ++r) {
        const float p = __expf(sc[nt][r] - mnew[r]);
        rs[r] += p;
        sP[wid][r + 8 * kh][nt * 16 + nl] = f2bf(p);
      }
    }
#pragma unroll
    for (int r = 0; r < 8; ++r) {
      rs[r] += __shfl_xor(rs[r], 1, 32);
      rs[r] += __shfl_xor(rs[r], 2, 32);
      rs[r] += __shfl_xor(rs[r], 4, 32);
      rs[r] += __shfl_xor(rs[r], 8, 32);
      lrow[r] = lrow[r] * alpha[r] + rs[r];
      mrow[r] = mnew[r];
    }
#pragma unroll
    for (int nt2 = 0; nt2 < 8; ++nt2)
#pragma unroll
      for (int r = 0; r < 8; ++r) o[nt2][r] *= alpha[r];

    // O += P @ V (2 k-steps over 64 keys, 8 output tiles of 16 dims)
#pragma unroll
    for (int ks2 = 0; ks2 < 2; ++ks2) {
      v16bf pf = lds_a_frag<72>((const unsigned short(*)[72])sP[wid], 0,
                                ks2 * 32, lane);
#pragma unroll
      for (int nt2 = 0; nt2 < 8; ++nt2)
        o[nt2] = wmma_bf16(
            pf, lds_bt_frag<72>(sV, nt2 * 16, ks2 * 32, lane), o[nt2]);
    }
    __syncthreads();
  }

#pragma unroll
  for (int nt2 = 0; nt2 < 8; ++nt2) {
#pragma unroll
    for (int r = 0; r < 8; ++r) {
      const int qg = q0 + wid * 16 + r + 8 * kh;
      Ob[((size_t)(b * S + qg) * NH + h) * HD + nt2 * 16 + nl] =
          f2bf(o[nt2][r] / lrow[r]);
    }
  }
}

// ---------------------------------------------------------------------------
// Host-side orchestration
// ---------------------------------------------------------------------------
extern "C" void kernel_launch(void* const* d_in, const int* in_sizes, int n_in,
                              void* d_out, int out_size, void* d_ws,
                              size_t ws_size, hipStream_t stream) {
  (void)in_sizes; (void)n_in; (void)out_size; (void)ws_size;
  const int B = 2, S = 2048, H = 2048, NH = 16, HD = 128, RD = 8;
  const int KVC = 512, QC = 1536, Dp = 160;
  const int M = B * S;

  const float* hidden = (const float*)d_in[0];
  const int* pos_ids = (const int*)d_in[2];
  const float* W_DKV = (const float*)d_in[3];  const float* b_DKV = (const float*)d_in[4];
  const float* W_DQ  = (const float*)d_in[5];  const float* b_DQ  = (const float*)d_in[6];
  const float* W_UK  = (const float*)d_in[7];  const float* b_UK  = (const float*)d_in[8];
  const float* W_UV  = (const float*)d_in[9];  const float* b_UV  = (const float*)d_in[10];
  const float* W_UQ  = (const float*)d_in[11]; const float* b_UQ  = (const float*)d_in[12];
  const float* W_KR  = (const float*)d_in[13]; const float* b_KR  = (const float*)d_in[14];
  const float* W_QR  = (const float*)d_in[15]; const float* b_QR  = (const float*)d_in[16];
  const float* W_O   = (const float*)d_in[17]; const float* b_O   = (const float*)d_in[18];
  const float* cosc  = (const float*)d_in[19];
  const float* sinc  = (const float*)d_in[20];
  float* out_f = (float*)d_out;

  char* base = (char*)d_ws;
  size_t off = 0;
  auto alloc = [&](size_t bytes) -> void* {
    void* p = base + off;
    off += (bytes + 255) & ~(size_t)255;
    return p;
  };
  typedef unsigned short us;
  us* hid_bf = (us*)alloc((size_t)M * H * 2);
  us* wdkvT = (us*)alloc((size_t)H * KVC * 2);   // [KVC,H]
  us* wdqT  = (us*)alloc((size_t)H * QC * 2);    // [QC,H]
  us* wukT  = (us*)alloc((size_t)KVC * NH * HD * 2);
  us* wuvT  = (us*)alloc((size_t)KVC * NH * HD * 2);
  us* wuqT  = (us*)alloc((size_t)QC * NH * HD * 2);
  us* wkrT  = (us*)alloc((size_t)H * NH * RD * 2);
  us* wqrT  = (us*)alloc((size_t)QC * NH * RD * 2);
  us* woT   = (us*)alloc((size_t)NH * HD * H * 2);
  us* ckv_bf = (us*)alloc((size_t)M * KVC * 2);
  us* cq_bf  = (us*)alloc((size_t)M * QC * 2);
  float* klat = (float*)alloc((size_t)M * NH * HD * 4);  // reused for attn-out
  float* qlat = (float*)alloc((size_t)M * NH * HD * 4);
  float* krop = (float*)alloc((size_t)M * NH * RD * 4);
  float* qrop = (float*)alloc((size_t)M * NH * RD * 4);
  us* Qp = (us*)alloc((size_t)B * NH * S * Dp * 2);
  us* Kp = (us*)alloc((size_t)B * NH * S * Dp * 2);
  us* vT_bf = (us*)alloc((size_t)M * NH * HD * 2);  // [B,NH,HD,S]
  us* ao_bf = (us*)klat;  // overlay: klat f32 dead after rope_pack(K)

  cast_f32_bf16_kernel<<<dim3((unsigned)((size_t)M * H / 256)), dim3(256), 0,
                         stream>>>(hidden, hid_bf, (size_t)M * H);
  auto castT = [&](const float* in, us* outp, int K, int N) {
    castT_f32_bf16_kernel<<<dim3((unsigned)((size_t)K * N / 256)), dim3(256),
                            0, stream>>>(in, outp, K, N);
  };
  castT(W_DKV, wdkvT, H, KVC);
  castT(W_DQ, wdqT, H, QC);
  castT(W_UK, wukT, KVC, NH * HD);
  castT(W_UV, wuvT, KVC, NH * HD);
  castT(W_UQ, wuqT, QC, NH * HD);
  castT(W_KR, wkrT, H, NH * RD);
  castT(W_QR, wqrT, QC, NH * RD);
  castT(W_O, woT, NH * HD, H);

  auto gemm = [&](const us* A, int lda, const us* BT, int ldbt,
                  const float* bias, float* Cf, us* Cb, int ldc, int vt,
                  int Mm, int Nn, int Kk) {
    gemm_bf16_kernel<<<dim3(Nn / 128, Mm / 128), dim3(256), 0, stream>>>(
        A, lda, BT, ldbt, bias, Cf, Cb, ldc, vt, Mm, Nn, Kk);
  };

  gemm(hid_bf, H, wdkvT, H, b_DKV, nullptr, ckv_bf, KVC, 0, M, KVC, H);
  gemm(hid_bf, H, wdqT, H, b_DQ, nullptr, cq_bf, QC, 0, M, QC, H);
  gemm(hid_bf, H, wkrT, H, b_KR, krop, nullptr, NH * RD, 0, M, NH * RD, H);
  gemm(ckv_bf, KVC, wukT, KVC, b_UK, klat, nullptr, NH * HD, 0, M, NH * HD, KVC);
  gemm(ckv_bf, KVC, wuvT, KVC, b_UV, nullptr, vT_bf, NH * HD, 1, M, NH * HD, KVC);
  gemm(cq_bf, QC, wuqT, QC, b_UQ, qlat, nullptr, NH * HD, 0, M, NH * HD, QC);
  gemm(cq_bf, QC, wqrT, QC, b_QR, qrop, nullptr, NH * RD, 0, M, NH * RD, QC);

  rope_pack_kernel<<<dim3(S, B), dim3(128), 0, stream>>>(klat, krop, cosc,
                                                         sinc, pos_ids, Kp);
  rope_pack_kernel<<<dim3(S, B), dim3(128), 0, stream>>>(qlat, qrop, cosc,
                                                         sinc, pos_ids, Qp);

  const float inv_scale = 1.0f / sqrtf((float)(HD + RD));
  mla_attn_kernel<<<dim3(S / 128, NH, B), dim3(256), 0, stream>>>(
      Qp, Kp, vT_bf, ao_bf, inv_scale);

  gemm(ao_bf, NH * HD, woT, H, b_O, out_f, nullptr, H, 0, M, H, NH * HD);
}
